// MambaBackbone_29076928593987
// MI455X (gfx1250) — compile-verified, matrix operand order checked by reading, not run on hardware
//
#include <hip/hip_runtime.h>

// ---------------- types ----------------
typedef __bf16 bf16_t;
typedef __attribute__((ext_vector_type(16))) __bf16 v16bf;
typedef __attribute__((ext_vector_type(8)))  __bf16 v8bf;
typedef __attribute__((ext_vector_type(8)))  float  v8f;
typedef int mb_v4i __attribute__((vector_size(16)));   // matches builtin proto

// ---------------- model constants ----------------
#define MB_B      8
#define MB_L      1024
#define MB_DM     768          // d_model
#define MB_DI     1536         // d_inner
#define MB_NST    64           // d_state
#define MB_PROJ   176          // dt_rank + 2*d_state
#define MB_RANK   48
#define MB_ROWS   (MB_B * MB_L)  // 8192

// ---------------- CDNA5 async-to-LDS path (hedged) ----------------
#if defined(__has_builtin)
#if __has_builtin(__builtin_amdgcn_global_load_async_to_lds_b128)
#define MB_USE_ASYNC_LDS 1
#endif
#endif
#ifndef MB_USE_ASYNC_LDS
#define MB_USE_ASYNC_LDS 0
#endif

// copy 16 bytes global -> LDS (async DMA when available)
__device__ __forceinline__ void copy16_g2l(const bf16_t* __restrict__ g,
                                           bf16_t* __restrict__ l) {
#if MB_USE_ASYNC_LDS
    __builtin_amdgcn_global_load_async_to_lds_b128(
        (__attribute__((address_space(1))) mb_v4i*)g,
        (__attribute__((address_space(3))) mb_v4i*)l, 0, 0);
#else
    *reinterpret_cast<v8bf*>(l) = *reinterpret_cast<const v8bf*>(g);
#endif
}

__device__ __forceinline__ void wait_g2l() {
#if MB_USE_ASYNC_LDS
#if __has_builtin(__builtin_amdgcn_s_wait_asynccnt)
    __builtin_amdgcn_s_wait_asynccnt(0);
#else
    asm volatile("s_wait_asynccnt 0x0" ::: "memory");
#endif
#endif
}

// ---------------- helpers ----------------
__device__ __forceinline__ float silu_f(float x) {
    return x * (1.0f / (1.0f + __expf(-x)));
}
__device__ __forceinline__ float softplus_f(float x) {
    return (x > 20.0f) ? x : __logf(1.0f + __expf(x));
}

// fp32 -> bf16 grid-stride convert
__global__ void cvt_f32_to_bf16(const float* __restrict__ src,
                                bf16_t* __restrict__ dst, size_t n) {
    size_t i = (size_t)blockIdx.x * blockDim.x + threadIdx.x;
    size_t stride = (size_t)gridDim.x * blockDim.x;
    for (; i < n; i += stride) dst[i] = (bf16_t)src[i];
}

// Load a 16x32 bf16 fragment from an LDS tile (row stride ldt elems), per the
// CDNA5 16-bit matrix VGPR layout:
//   lanes 0-15 : row = row0 + lane,    K = {0..7, 16..23}
//   lanes 16-31: row = row0 + lane-16, K = {8..15, 24..31}
__device__ __forceinline__ v16bf load_frag_lds(const bf16_t* base, int row0,
                                               int ldt) {
    int lane = threadIdx.x & 31;
    int r    = row0 + (lane & 15);
    int kk   = (lane >> 4) ? 8 : 0;
    const bf16_t* p = base + r * ldt + kk;
    v8bf lo = *reinterpret_cast<const v8bf*>(p);        // K +0..+7
    v8bf hi = *reinterpret_cast<const v8bf*>(p + 16);   // K +16..+23
    v16bf f;
#pragma unroll
    for (int i = 0; i < 8; ++i) { f[i] = lo[i]; f[i + 8] = hi[i]; }
    return f;
}

// ---------------- WMMA GEMM: C[M,N] = A[M,K] * B[N,K]^T ----------------
// A row-major (M,K) bf16; B row-major (N,K) bf16 (weight [out,in]);
// C row-major (M,N) fp32. Block = 256 threads = 8 waves.
// Block tile: 128 x (16*NT), K-step 32; A/B tiles double-buffered in LDS,
// filled with async-to-LDS DMA overlapped against the WMMA stream.
template <int NT>
__global__ void wmma_gemm_bf16(const bf16_t* __restrict__ A,
                               const bf16_t* __restrict__ B,
                               float* __restrict__ C,
                               int M, int N, int K) {
    constexpr int LDT = 40;            // LDS row stride (bf16), 80B = 16B-mult
    constexpr int BR  = NT * 16;       // B-tile rows
    __shared__ __align__(16) bf16_t As[2][128 * LDT];
    __shared__ __align__(16) bf16_t Bs[2][BR * LDT];

    const int tid  = threadIdx.x;
    const int lane = tid & 31;
    const int wave = tid >> 5;                   // 0..7
    const int m0 = blockIdx.y * 128;
    const int n0 = blockIdx.x * (16 * NT);

    auto load_tiles = [&](int buf, int k0) {
        // A tile: 128 rows x 32 bf16 = 512 16B-chunks -> 2 per thread
#pragma unroll
        for (int i = 0; i < 2; ++i) {
            int c = tid + 256 * i;
            int row = c >> 2, q = c & 3;
            copy16_g2l(A + (size_t)(m0 + row) * K + k0 + q * 8,
                       &As[buf][row * LDT + q * 8]);
        }
        // B tile: BR rows x 32 bf16 = BR*4 chunks -> 1 per thread
        if (tid < BR * 4) {
            int row = tid >> 2, q = tid & 3;
            copy16_g2l(B + (size_t)(n0 + row) * K + k0 + q * 8,
                       &Bs[buf][row * LDT + q * 8]);
        }
    };

    v8f acc[NT];
#pragma unroll
    for (int t = 0; t < NT; ++t) acc[t] = (v8f){0, 0, 0, 0, 0, 0, 0, 0};

    const int nk = K >> 5;
    load_tiles(0, 0);
    wait_g2l();
    __syncthreads();

    int buf = 0;
    for (int ki = 0; ki < nk; ++ki) {
        if (ki + 1 < nk) load_tiles(buf ^ 1, (ki + 1) << 5);   // overlap DMA

        v16bf a = load_frag_lds(&As[buf][0], wave * 16, LDT);
#pragma unroll
        for (int t = 0; t < NT; ++t) {
            v16bf b = load_frag_lds(&Bs[buf][0], t * 16, LDT);
            acc[t] = __builtin_amdgcn_wmma_f32_16x16x32_bf16(
                false, a, false, b, (short)0, acc[t], false, false);
        }
        wait_g2l();
        __syncthreads();
        buf ^= 1;
    }

    // C/D layout: lanes 0-15 -> rows m0+w*16+0..7, col n0+lane;
    //             lanes 16-31 -> rows +8..15, col n0+lane-16.
    int row = m0 + wave * 16 + ((lane >> 4) ? 8 : 0);
    int col = (lane & 15);
#pragma unroll
    for (int t = 0; t < NT; ++t) {
#pragma unroll
        for (int r = 0; r < 8; ++r) {
            C[(size_t)(row + r) * N + n0 + t * 16 + col] = acc[t][r];
        }
    }
}

// ---------------- causal depthwise conv (k=4) + SiLU -> u (bf16) ------------
// xz: [ROWS, 2*DI] fp32, xs = first DI columns; 4 channels per thread.
__global__ void conv_silu_kernel(const float* __restrict__ xz,
                                 const float* __restrict__ cw,   // [DI,4]
                                 const float* __restrict__ cb,   // [DI]
                                 bf16_t* __restrict__ u16) {
    const int DI4 = MB_DI / 4;
    size_t idx = (size_t)blockIdx.x * blockDim.x + threadIdx.x;
    const size_t total = (size_t)MB_ROWS * DI4;
    if (idx >= total) return;
    int d4 = (int)(idx % DI4);
    size_t row = idx / DI4;
    int l = (int)(row % MB_L);

    const float4* cw4 = (const float4*)cw;
    float4 w[4];                           // w[c] = 4 taps of channel d4*4+c
#pragma unroll
    for (int c = 0; c < 4; ++c) w[c] = cw4[d4 * 4 + c];
    float4 bias = ((const float4*)cb)[d4];
    float acc[4] = {bias.x, bias.y, bias.z, bias.w};

#pragma unroll
    for (int j = 0; j < 4; ++j) {
        int ls = l - 3 + j;
        if (ls >= 0) {
            float4 xv = ((const float4*)xz)[(row + (size_t)(ls - l)) *
                                                (2 * MB_DI / 4) + d4];
            acc[0] += xv.x * ((const float*)&w[0])[j];
            acc[1] += xv.y * ((const float*)&w[1])[j];
            acc[2] += xv.z * ((const float*)&w[2])[j];
            acc[3] += xv.w * ((const float*)&w[3])[j];
        }
    }
    size_t o = row * MB_DI + (size_t)d4 * 4;
#pragma unroll
    for (int c = 0; c < 4; ++c) u16[o + c] = (bf16_t)silu_f(acc[c]);
}

// ---------------- dt = softplus(dt_r @ W_dt^T + b_dt) ----------------------
__global__ void dt_proj_softplus(const float* __restrict__ proj,
                                 const float* __restrict__ Wdt,  // [DI,48]
                                 const float* __restrict__ bdt,  // [DI]
                                 float* __restrict__ dt) {
    __shared__ float ps[MB_RANK];
    size_t row = blockIdx.y;
    int d = blockIdx.x * blockDim.x + threadIdx.x;
    if (threadIdx.x < MB_RANK)
        ps[threadIdx.x] = proj[row * MB_PROJ + threadIdx.x];
    __syncthreads();
    float acc = bdt[d];
    const float* w = Wdt + (size_t)d * MB_RANK;
#pragma unroll
    for (int r = 0; r < MB_RANK; ++r) acc += ps[r] * w[r];
    dt[row * MB_DI + d] = softplus_f(acc);
}

// ---------------- selective scan + skip + gate, fused ----------------------
// One block = one batch x 256 channels; per-lane state h[64] and A-row[64]
// in VGPRs; B_t / C_t staged through LDS as float4 (ds_load_b128).
__global__ void mamba_scan_kernel(const float* __restrict__ dt,     // [ROWS,DI]
                                  const bf16_t* __restrict__ u16,   // [ROWS,DI]
                                  const float* __restrict__ proj,   // [ROWS,176]
                                  const float* __restrict__ xz,     // [ROWS,2*DI]
                                  const float* __restrict__ A_log,  // [DI,64]
                                  const float* __restrict__ Dp,     // [DI]
                                  bf16_t* __restrict__ yg16) {      // [ROWS,DI]
    int d = blockIdx.x * blockDim.x + threadIdx.x;   // channel
    int b = blockIdx.y;                              // batch

    float h[MB_NST];
    float Arow[MB_NST];
    const float* al = A_log + (size_t)d * MB_NST;
#pragma unroll
    for (int n = 0; n < MB_NST; ++n) {
        h[n] = 0.0f;
        Arow[n] = -__expf(al[n]);
    }
    float dskip = Dp[d];

    __shared__ __align__(16) float4 Bs4[MB_NST / 4];
    __shared__ __align__(16) float4 Cs4[MB_NST / 4];

    for (int l = 0; l < MB_L; ++l) {
        size_t row = (size_t)b * MB_L + l;
        if (threadIdx.x < MB_NST)
            ((float*)Bs4)[threadIdx.x] = proj[row * MB_PROJ + MB_RANK + threadIdx.x];
        else if (threadIdx.x < 2 * MB_NST)
            ((float*)Cs4)[threadIdx.x - MB_NST] =
                proj[row * MB_PROJ + MB_RANK + MB_NST + (threadIdx.x - MB_NST)];
        __syncthreads();

        float sdt = dt[row * MB_DI + d];
        float su  = (float)u16[row * MB_DI + d];
        float zz  = xz[row * (2 * MB_DI) + MB_DI + d];
        float dtu = sdt * su;
        float y = 0.0f;
#pragma unroll
        for (int n4 = 0; n4 < MB_NST / 4; ++n4) {
            float4 bb = Bs4[n4];
            float4 cc = Cs4[n4];
#pragma unroll
            for (int c = 0; c < 4; ++c) {
                int n = n4 * 4 + c;
                float bv = ((const float*)&bb)[c];
                float cv = ((const float*)&cc)[c];
                float dA = __expf(sdt * Arow[n]);
                h[n] = dA * h[n] + dtu * bv;
                y += h[n] * cv;
            }
        }
        float yv = (y + su * dskip) * silu_f(zz);
        yg16[row * MB_DI + d] = (bf16_t)yv;
        __syncthreads();
    }
}

// ---------------- host orchestration ----------------
extern "C" void kernel_launch(void* const* d_in, const int* in_sizes, int n_in,
                              void* d_out, int out_size, void* d_ws, size_t ws_size,
                              hipStream_t stream) {
    const float* x      = (const float*)d_in[0];
    const float* W_in   = (const float*)d_in[1];  // [2, 3072, 768]
    const float* conv_w = (const float*)d_in[2];  // [2, 1536, 4]
    const float* conv_b = (const float*)d_in[3];  // [2, 1536]
    const float* W_x    = (const float*)d_in[4];  // [2, 176, 1536]
    const float* W_dt   = (const float*)d_in[5];  // [2, 1536, 48]
    const float* b_dt   = (const float*)d_in[6];  // [2, 1536]
    const float* A_log  = (const float*)d_in[7];  // [2, 1536, 64]
    const float* Dp     = (const float*)d_in[8];  // [2, 1536]
    const float* W_out  = (const float*)d_in[9];  // [2, 768, 1536]
    float* out = (float*)d_out;

    const size_t ROWS = MB_ROWS;
    char* ws = (char*)d_ws;
    auto aln = [](size_t v) { return (v + 255) & ~(size_t)255; };
    size_t off = 0;
    float*  xz   = (float*) (ws + off); off = aln(off + ROWS * 2 * MB_DI * sizeof(float));
    bf16_t* x16  = (bf16_t*)(ws + off); off = aln(off + ROWS * MB_DM * sizeof(bf16_t));
    bf16_t* w16  = (bf16_t*)(ws + off); off = aln(off + (size_t)2 * MB_DI * MB_DM * sizeof(bf16_t));
    bf16_t* u16  = (bf16_t*)(ws + off); off = aln(off + ROWS * MB_DI * sizeof(bf16_t));
    float*  proj = (float*) (ws + off); off = aln(off + ROWS * MB_PROJ * sizeof(float));
    float*  dt   = (float*) (ws + off); off = aln(off + ROWS * MB_DI * sizeof(float));
    bf16_t* yg   = (bf16_t*)(ws + off); off = aln(off + ROWS * MB_DI * sizeof(bf16_t));
    float*  y1   = (float*) (ws + off); off = aln(off + ROWS * MB_DM * sizeof(float));
    (void)ws_size; (void)in_sizes; (void)n_in; (void)out_size;

    for (int layer = 0; layer < 2; ++layer) {
        const float* xin  = (layer == 0) ? x  : y1;
        float*       yout = (layer == 0) ? y1 : out;

        // 1) bf16 conversions for input-proj GEMM
        cvt_f32_to_bf16<<<1024, 256, 0, stream>>>(xin, x16, ROWS * MB_DM);
        cvt_f32_to_bf16<<<1024, 256, 0, stream>>>(
            W_in + (size_t)layer * 2 * MB_DI * MB_DM, w16, (size_t)2 * MB_DI * MB_DM);

        // 2) xz = x @ W_in^T   (8192 x 3072, K=768)
        wmma_gemm_bf16<4><<<dim3(2 * MB_DI / 64, ROWS / 128), 256, 0, stream>>>(
            x16, w16, xz, (int)ROWS, 2 * MB_DI, MB_DM);

        // 3) causal depthwise conv + SiLU -> u (bf16)
        conv_silu_kernel<<<(unsigned)((ROWS * (MB_DI / 4) + 255) / 256), 256, 0,
                          stream>>>(
            xz, conv_w + (size_t)layer * MB_DI * 4, conv_b + (size_t)layer * MB_DI, u16);

        // 4) proj = u @ W_x^T  (8192 x 176, K=1536)
        cvt_f32_to_bf16<<<1024, 256, 0, stream>>>(
            W_x + (size_t)layer * MB_PROJ * MB_DI, w16, (size_t)MB_PROJ * MB_DI);
        wmma_gemm_bf16<1><<<dim3(MB_PROJ / 16, ROWS / 128), 256, 0, stream>>>(
            u16, w16, proj, (int)ROWS, MB_PROJ, MB_DI);

        // 5) dt = softplus(dt_r @ W_dt^T + b_dt)   (K=48, VALU)
        dt_proj_softplus<<<dim3(MB_DI / 256, (unsigned)ROWS), 256, 0, stream>>>(
            proj, W_dt + (size_t)layer * MB_DI * MB_RANK,
            b_dt + (size_t)layer * MB_DI, dt);

        // 6) selective scan + skip + gate (fused) -> yg (bf16)
        mamba_scan_kernel<<<dim3(MB_DI / 256, MB_B), 256, 0, stream>>>(
            dt, u16, proj, xz,
            A_log + (size_t)layer * MB_DI * MB_NST,
            Dp + (size_t)layer * MB_DI, yg);

        // 7) out = yg @ W_out^T  (8192 x 768, K=1536)
        cvt_f32_to_bf16<<<1024, 256, 0, stream>>>(
            W_out + (size_t)layer * MB_DM * MB_DI, w16, (size_t)MB_DM * MB_DI);
        wmma_gemm_bf16<4><<<dim3(MB_DM / 64, ROWS / 128), 256, 0, stream>>>(
            yg, w16, yout, (int)ROWS, MB_DM, MB_DI);
    }
}
